// RGCN_25975962206903
// MI455X (gfx1250) — matrix operand order repaired and checked by reading, not compiled, not run on hardware
//
#include <hip/hip_runtime.h>
#include <hip/hip_bf16.h>

// Problem sizes (match reference).
constexpr int Bz = 16;   // batch
constexpr int Rr = 5;    // relations
constexpr int Nn = 512;  // nodes
constexpr int Dd = 512;  // hidden dim

typedef __attribute__((ext_vector_type(16))) __bf16 bf16x16;
typedef __attribute__((ext_vector_type(8)))  __bf16 bf16x8;
typedef __attribute__((ext_vector_type(8)))  float  f32x8;
typedef __attribute__((ext_vector_type(4)))  float  f32x4;

// ---------------------------------------------------------------------------
// WMMA helper: D = A(16x32 bf16) * B(32x16 bf16) + C(16x16 f32)
// ---------------------------------------------------------------------------
__device__ __forceinline__ f32x8 wmma_bf16(bf16x16 a, bf16x16 b, f32x8 c) {
  return __builtin_amdgcn_wmma_f32_16x16x32_bf16(
      /*neg_a=*/false, a, /*neg_b=*/false, b,
      /*c_mod=*/(short)0, c, /*reuse_a=*/false, /*reuse_b=*/false);
}

// A operand (16x32 tile, row-major bf16, leading dim ld).
// Lane L: row = L%16; K in [off,off+8) and [16+off,16+off+8), off=8*(L>=16).
__device__ __forceinline__ bf16x16 load_a_bf16(const __bf16* __restrict__ p,
                                               int ld, int lane) {
  const __bf16* row = p + (lane & 15) * ld;
  const int off = (lane >> 4) << 3;            // 0 or 8
  bf16x8 lo = *(const bf16x8*)(row + off);
  bf16x8 hi = *(const bf16x8*)(row + 16 + off);
  bf16x16 a;
#pragma unroll
  for (int i = 0; i < 8; ++i) { a[i] = lo[i]; a[8 + i] = hi[i]; }
  return a;
}

// Same A layout but source is fp32 (adjacency, converted in-register).
__device__ __forceinline__ bf16x16 load_a_f32(const float* __restrict__ p,
                                              int ld, int lane) {
  const float* row = p + (lane & 15) * ld;
  const int off = (lane >> 4) << 3;
  f32x4 a0 = *(const f32x4*)(row + off);
  f32x4 a1 = *(const f32x4*)(row + off + 4);
  f32x4 b0 = *(const f32x4*)(row + 16 + off);
  f32x4 b1 = *(const f32x4*)(row + 16 + off + 4);
  bf16x16 a;
#pragma unroll
  for (int i = 0; i < 4; ++i) {
    a[i]      = (__bf16)a0[i];
    a[4 + i]  = (__bf16)a1[i];
    a[8 + i]  = (__bf16)b0[i];
    a[12 + i] = (__bf16)b1[i];
  }
  return a;
}

// B operand (32x16 tile). Source stored as [n][k] row-major (ld = k-stride).
// Lane L: col n = L%16; K in [off,off+8) and [16+off,16+off+8), off=8*(L>=16).
__device__ __forceinline__ bf16x16 load_b_bf16(const __bf16* __restrict__ p,
                                               int ld, int lane) {
  const __bf16* row = p + (lane & 15) * ld;
  const int off = (lane >> 4) << 3;
  bf16x8 lo = *(const bf16x8*)(row + off);
  bf16x8 hi = *(const bf16x8*)(row + 16 + off);
  bf16x16 b;
#pragma unroll
  for (int i = 0; i < 8; ++i) { b[i] = lo[i]; b[8 + i] = hi[i]; }
  return b;
}

// ---------------------------------------------------------------------------
// Pass 0: fp32 -> bf16 conversion (h, W_rel, W_hid, W_gate)
// ---------------------------------------------------------------------------
__global__ void __launch_bounds__(256)
cvt_f32_bf16(const float* __restrict__ s, __bf16* __restrict__ d, int n) {
  int i = blockIdx.x * blockDim.x + threadIdx.x;
  int stride = gridDim.x * blockDim.x;
  for (; i < n; i += stride) d[i] = (__bf16)s[i];
}

// ---------------------------------------------------------------------------
// Pass 2: h_i[b,r,m,o] = mask[b,m] * (h[b,m,:] . W_rel[r,o,:] + b_rel[r,o])
//         stored TRANSPOSED as hIT[b,r,o,m] (bf16).
//         One wave per 32x32 macro-tile (2x2 register blocking).
// ---------------------------------------------------------------------------
__global__ void __launch_bounds__(256)
rel_gemm(const __bf16* __restrict__ hB, const __bf16* __restrict__ WrelB,
         const float* __restrict__ b_rel, const float* __restrict__ node_mask,
         __bf16* __restrict__ hIT) {
  const int lane = threadIdx.x & 31;
  const int wid  = blockIdx.x * (blockDim.x >> 5) + (threadIdx.x >> 5);
  const int ot = wid & 15;            // 16 o-macro-tiles
  const int nt = (wid >> 4) & 15;     // 16 n-macro-tiles
  const int br = wid >> 8;
  const int r  = br % Rr;
  const int b  = br / Rr;
  const int n0 = nt * 32, o0 = ot * 32;

  const __bf16* Abase = hB + (b * Nn + n0) * Dd;
  const __bf16* Bbase = WrelB + (r * Dd + o0) * Dd;

  f32x8 acc00 = {}, acc01 = {}, acc10 = {}, acc11 = {};
#pragma unroll 2
  for (int k0 = 0; k0 < Dd; k0 += 32) {
    bf16x16 a0 = load_a_bf16(Abase + k0,           Dd, lane);
    bf16x16 a1 = load_a_bf16(Abase + 16 * Dd + k0, Dd, lane);
    bf16x16 b0 = load_b_bf16(Bbase + k0,           Dd, lane);
    bf16x16 b1 = load_b_bf16(Bbase + 16 * Dd + k0, Dd, lane);
    acc00 = wmma_bf16(a0, b0, acc00);
    acc01 = wmma_bf16(a0, b1, acc01);
    acc10 = wmma_bf16(a1, b0, acc10);
    acc11 = wmma_bf16(a1, b1, acc11);
  }

  const int col = lane & 15;
  const int hh  = (lane >> 4) << 3;   // 0 or 8
  f32x8 accs[2][2] = {{acc00, acc01}, {acc10, acc11}};
#pragma unroll
  for (int i = 0; i < 2; ++i) {
#pragma unroll
    for (int j = 0; j < 2; ++j) {
      const float bias = b_rel[r * Dd + o0 + 16 * j + col];
      bf16x8 outv;
#pragma unroll
      for (int v = 0; v < 8; ++v) {
        const float mk = node_mask[b * Nn + n0 + 16 * i + hh + v];
        outv[v] = (__bf16)((accs[i][j][v] + bias) * mk);
      }
      // transposed store: contiguous over the 8 rows m = n0+16i+hh+v
      *(bf16x8*)(hIT + (((b * Rr + r) * Dd + o0 + 16 * j + col) * Nn +
                        n0 + 16 * i + hh)) = outv;
    }
  }
}

// ---------------------------------------------------------------------------
// Pass 3: update[b,n,o] = sum_r adj[b,r,n,:] . h_i[b,r,:,o]
//                       + mask[b,n] * (h[b,n,:] . W_hid[o,:] + b_hid[o])
//         32x32 macro-tile per wave; adjacency fp32->bf16 in-register.
// ---------------------------------------------------------------------------
__global__ void __launch_bounds__(256)
msg_update(const float* __restrict__ adj, const __bf16* __restrict__ hIT,
           const __bf16* __restrict__ hB, const __bf16* __restrict__ WhidB,
           const float* __restrict__ b_hid, const float* __restrict__ node_mask,
           float* __restrict__ updF, __bf16* __restrict__ updB) {
  const int lane = threadIdx.x & 31;
  const int wid  = blockIdx.x * (blockDim.x >> 5) + (threadIdx.x >> 5);
  const int ot = wid & 15;
  const int nt = (wid >> 4) & 15;
  const int b  = wid >> 8;
  const int n0 = nt * 32, o0 = ot * 32;

  f32x8 m00 = {}, m01 = {}, m10 = {}, m11 = {};
  for (int r = 0; r < Rr; ++r) {
    const float*  Ab = adj + (((b * Rr + r) * Nn + n0) * Nn);
    const __bf16* Bb = hIT + (((b * Rr + r) * Dd + o0) * Nn);
    for (int k0 = 0; k0 < Nn; k0 += 32) {
      bf16x16 a0 = load_a_f32(Ab + k0,           Nn, lane);
      bf16x16 a1 = load_a_f32(Ab + 16 * Nn + k0, Nn, lane);
      bf16x16 b0 = load_b_bf16(Bb + k0,           Nn, lane);
      bf16x16 b1 = load_b_bf16(Bb + 16 * Nn + k0, Nn, lane);
      m00 = wmma_bf16(a0, b0, m00);
      m01 = wmma_bf16(a0, b1, m01);
      m10 = wmma_bf16(a1, b0, m10);
      m11 = wmma_bf16(a1, b1, m11);
    }
  }

  f32x8 h00 = {}, h01 = {}, h10 = {}, h11 = {};
  const __bf16* Ah = hB + (b * Nn + n0) * Dd;
  const __bf16* Bh = WhidB + o0 * Dd;
#pragma unroll 2
  for (int k0 = 0; k0 < Dd; k0 += 32) {
    bf16x16 a0 = load_a_bf16(Ah + k0,           Dd, lane);
    bf16x16 a1 = load_a_bf16(Ah + 16 * Dd + k0, Dd, lane);
    bf16x16 b0 = load_b_bf16(Bh + k0,           Dd, lane);
    bf16x16 b1 = load_b_bf16(Bh + 16 * Dd + k0, Dd, lane);
    h00 = wmma_bf16(a0, b0, h00);
    h01 = wmma_bf16(a0, b1, h01);
    h10 = wmma_bf16(a1, b0, h10);
    h11 = wmma_bf16(a1, b1, h11);
  }

  const int col = lane & 15;
  const int hh  = (lane >> 4) << 3;
  f32x8 accM[2][2] = {{m00, m01}, {m10, m11}};
  f32x8 accH[2][2] = {{h00, h01}, {h10, h11}};
#pragma unroll
  for (int i = 0; i < 2; ++i) {
#pragma unroll
    for (int j = 0; j < 2; ++j) {
      const float bh = b_hid[o0 + 16 * j + col];
#pragma unroll
      for (int v = 0; v < 8; ++v) {
        const int row = n0 + 16 * i + hh + v;
        const float mk = node_mask[b * Nn + row];
        const float u  = accM[i][j][v] + (accH[i][j][v] + bh) * mk;
        const int idx  = (b * Nn + row) * Dd + o0 + 16 * j + col;
        updF[idx] = u;
        updB[idx] = (__bf16)u;
      }
    }
  }
}

// ---------------------------------------------------------------------------
// Pass 4: gate = sigmoid([update,h] @ W_gate^T + b_gate) * mask
//         out  = gate * tanh(update) + (1-gate) * h
//         32x32 macro-tile per wave.
// ---------------------------------------------------------------------------
__global__ void __launch_bounds__(256)
gate_out(const __bf16* __restrict__ updB, const __bf16* __restrict__ hB,
         const __bf16* __restrict__ WgateB, const float* __restrict__ b_gate,
         const float* __restrict__ node_mask, const float* __restrict__ updF,
         const float* __restrict__ hF, float* __restrict__ out) {
  const int lane = threadIdx.x & 31;
  const int wid  = blockIdx.x * (blockDim.x >> 5) + (threadIdx.x >> 5);
  const int ot = wid & 15;
  const int nt = (wid >> 4) & 15;
  const int b  = wid >> 8;
  const int n0 = nt * 32, o0 = ot * 32;

  const __bf16* A1 = updB + (b * Nn + n0) * Dd;   // update half of concat
  const __bf16* A2 = hB   + (b * Nn + n0) * Dd;   // h half of concat
  const __bf16* Bg = WgateB + o0 * (2 * Dd);

  f32x8 g00 = {}, g01 = {}, g10 = {}, g11 = {};
#pragma unroll 2
  for (int k0 = 0; k0 < Dd; k0 += 32) {
    bf16x16 a0 = load_a_bf16(A1 + k0,           Dd, lane);
    bf16x16 a1 = load_a_bf16(A1 + 16 * Dd + k0, Dd, lane);
    bf16x16 b0 = load_b_bf16(Bg + k0,                2 * Dd, lane);
    bf16x16 b1 = load_b_bf16(Bg + 16 * (2 * Dd) + k0, 2 * Dd, lane);
    g00 = wmma_bf16(a0, b0, g00);
    g01 = wmma_bf16(a0, b1, g01);
    g10 = wmma_bf16(a1, b0, g10);
    g11 = wmma_bf16(a1, b1, g11);
  }
#pragma unroll 2
  for (int k0 = 0; k0 < Dd; k0 += 32) {
    bf16x16 a0 = load_a_bf16(A2 + k0,           Dd, lane);
    bf16x16 a1 = load_a_bf16(A2 + 16 * Dd + k0, Dd, lane);
    bf16x16 b0 = load_b_bf16(Bg + Dd + k0,                2 * Dd, lane);
    bf16x16 b1 = load_b_bf16(Bg + Dd + 16 * (2 * Dd) + k0, 2 * Dd, lane);
    g00 = wmma_bf16(a0, b0, g00);
    g01 = wmma_bf16(a0, b1, g01);
    g10 = wmma_bf16(a1, b0, g10);
    g11 = wmma_bf16(a1, b1, g11);
  }

  const int col = lane & 15;
  const int hh  = (lane >> 4) << 3;
  f32x8 accs[2][2] = {{g00, g01}, {g10, g11}};
#pragma unroll
  for (int i = 0; i < 2; ++i) {
#pragma unroll
    for (int j = 0; j < 2; ++j) {
      const float bg = b_gate[o0 + 16 * j + col];
#pragma unroll
      for (int v = 0; v < 8; ++v) {
        const int row = n0 + 16 * i + hh + v;
        const float mk = node_mask[b * Nn + row];
        const float g  = mk / (1.0f + __expf(-(accs[i][j][v] + bg)));
        const int idx  = (b * Nn + row) * Dd + o0 + 16 * j + col;
        const float u  = updF[idx];
        const float hv = hF[idx];
        out[idx] = g * tanhf(u) + (1.0f - g) * hv;
      }
    }
  }
}

// ---------------------------------------------------------------------------
extern "C" void kernel_launch(void* const* d_in, const int* in_sizes, int n_in,
                              void* d_out, int out_size, void* d_ws, size_t ws_size,
                              hipStream_t stream) {
  const float* h      = (const float*)d_in[0];
  const float* adj    = (const float*)d_in[1];
  const float* mask   = (const float*)d_in[2];
  const float* W_rel  = (const float*)d_in[3];
  const float* b_rel  = (const float*)d_in[4];
  const float* W_hid  = (const float*)d_in[5];
  const float* b_hid  = (const float*)d_in[6];
  const float* W_gate = (const float*)d_in[7];
  const float* b_gate = (const float*)d_in[8];
  float* out = (float*)d_out;

  // Workspace carve-up (bf16 staging + fp32 update), ~76 MB total.
  char* p = (char*)d_ws;
  __bf16* hB     = (__bf16*)p; p += (size_t)Bz * Nn * Dd * 2;        //  8.4 MB
  __bf16* WrelB  = (__bf16*)p; p += (size_t)Rr * Dd * Dd * 2;        //  2.6 MB
  __bf16* WhidB  = (__bf16*)p; p += (size_t)Dd * Dd * 2;             //  0.5 MB
  __bf16* WgateB = (__bf16*)p; p += (size_t)Dd * 2 * Dd * 2;         //  1.0 MB
  __bf16* hIT    = (__bf16*)p; p += (size_t)Bz * Rr * Dd * Nn * 2;   // 41.9 MB
  __bf16* updB   = (__bf16*)p; p += (size_t)Bz * Nn * Dd * 2;        //  8.4 MB
  float*  updF   = (float*)p;  p += (size_t)Bz * Nn * Dd * 4;        // 16.8 MB
  if ((size_t)(p - (char*)d_ws) > ws_size) return;

  // Pass 0: bf16 staging of reused tensors.
  cvt_f32_bf16<<<2048, 256, 0, stream>>>(h,      hB,     Bz * Nn * Dd);
  cvt_f32_bf16<<<1024, 256, 0, stream>>>(W_rel,  WrelB,  Rr * Dd * Dd);
  cvt_f32_bf16<<<512,  256, 0, stream>>>(W_hid,  WhidB,  Dd * Dd);
  cvt_f32_bf16<<<512,  256, 0, stream>>>(W_gate, WgateB, Dd * 2 * Dd);

  // Pass 2: per-relation linear, masked, stored transposed. 32x32 tiles.
  {
    const int waves = Bz * Rr * (Nn / 32) * (Dd / 32);  // 20480
    rel_gemm<<<waves / 8, 256, 0, stream>>>(hB, WrelB, b_rel, mask, hIT);
  }
  // Pass 3: adjacency message GEMM + hidden GEMM -> update. 32x32 tiles.
  {
    const int waves = Bz * (Nn / 32) * (Dd / 32);       // 4096
    msg_update<<<waves / 8, 256, 0, stream>>>(adj, hIT, hB, WhidB, b_hid, mask,
                                              updF, updB);
  }
  // Pass 4: gate GEMM + elementwise blend -> output. 32x32 tiles.
  {
    const int waves = Bz * (Nn / 32) * (Dd / 32);       // 4096
    gate_out<<<waves / 8, 256, 0, stream>>>(updB, hB, WgateB, b_gate, mask,
                                            updF, h, out);
  }
}
